// UVMB_78932908966036
// MI455X (gfx1250) — matrix operand order repaired and checked by reading, not compiled, not run on hardware
//
#include <hip/hip_runtime.h>
#include <hip/hip_bf16.h>
#include <math.h>

// ---------------------------------------------------------------------------
// Types for CDNA5 WMMA
// ---------------------------------------------------------------------------
typedef __attribute__((ext_vector_type(16))) __bf16        v16bf;
typedef __attribute__((ext_vector_type(2)))  __bf16        v2bf;
typedef __attribute__((ext_vector_type(8)))  float         v8f;
typedef __attribute__((ext_vector_type(4)))  float         v4f;
typedef __attribute__((ext_vector_type(4)))  unsigned int  v4u;
// The async-copy builtin's pointer parameter type (from hipcc diagnostics):
typedef int v4i_vs __attribute__((vector_size(4 * sizeof(int))));

#if __has_builtin(__builtin_amdgcn_global_load_async_to_lds_b128) && \
    __has_builtin(__builtin_amdgcn_s_wait_asynccnt)
#define HAVE_ASYNC_LDS 1
#else
#define HAVE_ASYNC_LDS 0
#endif

// ---------------------------------------------------------------------------
// Helpers
// ---------------------------------------------------------------------------
__device__ __forceinline__ float sigm(float x) { return 1.f / (1.f + __expf(-x)); }
__device__ __forceinline__ float siluf(float x) { return x * sigm(x); }
__device__ __forceinline__ float softplusf(float x) {
    return (x > 20.f) ? x : log1pf(__expf(x));
}
// fp32 -> bf16: let the backend lower fptrunc to the native cvt instruction.
__device__ __forceinline__ __bf16 cvtbf(float f) {
#if __has_builtin(__builtin_amdgcn_cvt_pk_bf16_f32)
    v2bf r = __builtin_amdgcn_cvt_pk_bf16_f32(f, f);
    return r[0];
#else
    return (__bf16)f;
#endif
}

// ---------------------------------------------------------------------------
// Generic 3x3 conv, pad 1, NCHW on 64x64 images; optional ReLU / residual.
// ---------------------------------------------------------------------------
__global__ void conv3x3_kernel(const float* __restrict__ in,
                               const float* __restrict__ wgt,
                               const float* __restrict__ bias,
                               const float* __restrict__ residual,
                               float* __restrict__ out,
                               int Bn, int IC, int OC, int do_relu) {
    int idx = blockIdx.x * blockDim.x + threadIdx.x;
    int total = Bn * OC * 64 * 64;
    if (idx >= total) return;
    int xp = idx & 63;
    int yp = (idx >> 6) & 63;
    int oc = (idx >> 12) % OC;
    int b  = idx / (OC << 12);
    float acc = bias[oc];
    for (int ic = 0; ic < IC; ++ic) {
        const float* ip = in + ((size_t)(b * IC + ic) << 12);
        const float* wp = wgt + (size_t)(oc * IC + ic) * 9;
        #pragma unroll
        for (int ky = 0; ky < 3; ++ky) {
            int yy = yp + ky - 1;
            if (yy < 0 || yy > 63) continue;
            #pragma unroll
            for (int kx = 0; kx < 3; ++kx) {
                int xx = xp + kx - 1;
                if (xx < 0 || xx > 63) continue;
                acc += ip[yy * 64 + xx] * wp[ky * 3 + kx];
            }
        }
    }
    if (residual) acc += residual[idx];
    if (do_relu) acc = fmaxf(acc, 0.f);
    out[idx] = acc;
}

// ---------------------------------------------------------------------------
// LayerNorm over last axis of size 3
// ---------------------------------------------------------------------------
__global__ void layernorm_c3_kernel(const float* __restrict__ in,
                                    const float* __restrict__ g,
                                    const float* __restrict__ bb,
                                    float* __restrict__ out, int total) {
    int i = blockIdx.x * blockDim.x + threadIdx.x;
    if (i >= total) return;
    float a0 = in[i * 3 + 0], a1 = in[i * 3 + 1], a2 = in[i * 3 + 2];
    float m  = (a0 + a1 + a2) * (1.f / 3.f);
    float d0 = a0 - m, d1 = a1 - m, d2 = a2 - m;
    float v  = (d0 * d0 + d1 * d1 + d2 * d2) * (1.f / 3.f);
    float r  = rsqrtf(v + 1e-5f);
    out[i * 3 + 0] = d0 * r * g[0] + bb[0];
    out[i * 3 + 1] = d1 * r * g[1] + bb[1];
    out[i * 3 + 2] = d2 * r * g[2] + bb[2];
}

// ---------------------------------------------------------------------------
// Small mamba (m1/m2): d_model=3, d_inner=6, d_state=16, dt_rank=1, L=4096
// ---------------------------------------------------------------------------
__global__ void sm_inproj_kernel(const float* __restrict__ xs,
                                 const float* __restrict__ w,  // (12,3)
                                 float* __restrict__ xz, int total) {
    int i = blockIdx.x * blockDim.x + threadIdx.x;
    if (i >= total) return;
    float a0 = xs[i * 3], a1 = xs[i * 3 + 1], a2 = xs[i * 3 + 2];
    #pragma unroll
    for (int o = 0; o < 12; ++o)
        xz[(size_t)i * 12 + o] = w[o * 3] * a0 + w[o * 3 + 1] * a1 + w[o * 3 + 2] * a2;
}

// causal depthwise conv1d(width 4) + bias + silu. Row width of xz is 2*Din.
__global__ void dwconv_silu_kernel(const float* __restrict__ xz,
                                   const float* __restrict__ cw,
                                   const float* __restrict__ cb,
                                   float* __restrict__ xh,
                                   int Bn, int Lr, int Din) {
    int i = blockIdx.x * blockDim.x + threadIdx.x;
    int total = Bn * Lr * Din;
    if (i >= total) return;
    int d = i % Din;
    int l = (i / Din) % Lr;
    int b = i / (Din * Lr);
    float acc = cb[d];
    #pragma unroll
    for (int k = 0; k < 4; ++k) {
        int ll = l - 3 + k;
        if (ll >= 0)
            acc += cw[d * 4 + k] * xz[((size_t)(b * Lr + ll)) * (2 * Din) + d];
    }
    xh[i] = siluf(acc);
}

__global__ void sm_xproj_kernel(const float* __restrict__ xh,
                                const float* __restrict__ w,  // (33,6)
                                float* __restrict__ xdbl, int total) {
    int i = blockIdx.x * blockDim.x + threadIdx.x;
    if (i >= total) return;
    float a[6];
    #pragma unroll
    for (int d = 0; d < 6; ++d) a[d] = xh[(size_t)i * 6 + d];
    for (int o = 0; o < 33; ++o) {
        float s = 0.f;
        #pragma unroll
        for (int d = 0; d < 6; ++d) s += w[o * 6 + d] * a[d];
        xdbl[(size_t)i * 33 + o] = s;
    }
}

// One wave per (b,d): lanes 0..15 carry state n, sequential over L.
__global__ void sm_scan_kernel(const float* __restrict__ xdbl,
                               const float* __restrict__ xh,
                               const float* __restrict__ dtw,
                               const float* __restrict__ dtb,
                               const float* __restrict__ alog,
                               const float* __restrict__ Dp,
                               float* __restrict__ yout,
                               int Lr) {
    int bd = blockIdx.x;
    int d = bd % 6, b = bd / 6;
    int lane = threadIdx.x;
    bool act = lane < 16;
    int n = lane & 15;
    float A  = act ? -__expf(alog[d * 16 + n]) : 0.f;
    float dw = dtw[d], db = dtb[d], Dd = Dp[d];
    float h = 0.f;
    for (int l = 0; l < Lr; ++l) {
        size_t row = (size_t)b * Lr + l;
        float dt    = xdbl[row * 33];
        float delta = softplusf(dt * dw + db);
        float xv    = xh[row * 6 + d];
        float contrib = 0.f;
        if (act) {
            float Bv = xdbl[row * 33 + 1 + n];
            float Cv = xdbl[row * 33 + 17 + n];
            h = __expf(delta * A) * h + delta * Bv * xv;
            contrib = h * Cv;
        }
        #pragma unroll
        for (int off = 16; off > 0; off >>= 1)
            contrib += __shfl_down(contrib, off, 32);
        if (lane == 0) yout[row * 6 + d] = contrib + xv * Dd;
    }
}

__global__ void sm_gate_out_kernel(const float* __restrict__ y,
                                   const float* __restrict__ xz,
                                   const float* __restrict__ ow,  // (3,6)
                                   float* __restrict__ out, int total) {
    int i = blockIdx.x * blockDim.x + threadIdx.x;
    if (i >= total) return;
    float g[6];
    #pragma unroll
    for (int d = 0; d < 6; ++d) {
        float z = xz[(size_t)i * 12 + 6 + d];
        g[d] = y[(size_t)i * 6 + d] * siluf(z);
    }
    #pragma unroll
    for (int o = 0; o < 3; ++o) {
        float s = 0.f;
        #pragma unroll
        for (int d = 0; d < 6; ++d) s += ow[o * 6 + d] * g[d];
        out[(size_t)i * 3 + o] = s;
    }
}

// ---------------------------------------------------------------------------
// bf16 WMMA GEMM:  C[M,N] = A[M,K] * W[N,K]^T (+bias)
//   aMode 0: A row-major, row stride aStride (fully in-bounds, vector loads)
//   aMode 1: m3 in_proj gather: row r=(b*3+l), elem = src[(b*4096+k)*3 + l]
// Preconditions: K % 64 == 0, M % 16 == 0 (kills all K/M guards).
// Block: 128 thr (4 waves), tile 16(M) x 128(N), K-step 64.
// W tile: async-DMA'd raw fp32 into LDS (ASYNCcnt), converted with native
// bf16 fptrunc at fragment build. 4 WMMAs per K-stage per wave.
// ---------------------------------------------------------------------------
#define GEMM_NT 128
#define GEMM_KT 64

__global__ __launch_bounds__(128)
void gemm_bf16_wmma_kernel(const float* __restrict__ A, int aMode, int aStride,
                           const float* __restrict__ Wt,
                           const float* __restrict__ bias,
                           float* __restrict__ Cout,
                           int M, int N, int K) {
    __shared__ __bf16 As[16][GEMM_KT + 8];              // 72 bf16/row = 144B (16B-aligned)
    __shared__ float  WsF[GEMM_NT][GEMM_KT + 4];        // 68 f32/row = 272B (16B-aligned)

    int tid  = threadIdx.x;   // 0..127
    int wave = tid >> 5;      // 0..3
    int lane = tid & 31;
    int m0     = blockIdx.y * 16;
    int nBlock = blockIdx.x * GEMM_NT;

    v8f acc[2] = {{}, {}};
    int kTiles = K / GEMM_KT;

    for (int kt = 0; kt < kTiles; ++kt) {
        int kBase = kt * GEMM_KT;

        // ---- stage A tile (16x64) -> bf16, no guards ----
        if (aMode == 0) {
            #pragma unroll
            for (int it = 0; it < 2; ++it) {
                int idx = tid + it * 128;             // 0..255 float4s
                int r = idx >> 4, k4 = idx & 15;
                v4f v = *(const v4f*)&A[(size_t)(m0 + r) * aStride + kBase + k4 * 4];
                __bf16* dst = &As[r][k4 * 4];
                dst[0] = cvtbf(v[0]); dst[1] = cvtbf(v[1]);
                dst[2] = cvtbf(v[2]); dst[3] = cvtbf(v[3]);
            }
        } else {
            #pragma unroll
            for (int it = 0; it < 8; ++it) {
                int idx = tid + it * 128;             // 0..1023 elems
                int r = idx >> 6, k = idx & 63;
                int gm = m0 + r;
                int b = gm / 3, l = gm - b * 3;
                As[r][k] = cvtbf(A[((size_t)b * 4096 + kBase + k) * 3 + l]);
            }
        }

        // ---- stage W tile (128x64) raw fp32 via async DMA to LDS ----
        #pragma unroll
        for (int it = 0; it < 16; ++it) {
            int idx = tid + it * 128;                 // 0..2047 float4s
            int r = idx >> 4, k4 = idx & 15;
            int gn = nBlock + r;
            if (gn < N) {
                const float* gp = &Wt[(size_t)gn * K + kBase + k4 * 4];
                float*       lp = &WsF[r][k4 * 4];
#if HAVE_ASYNC_LDS
                __builtin_amdgcn_global_load_async_to_lds_b128(
                    (v4i_vs*)gp, (v4i_vs*)lp, 0, 0);
#else
                *(v4f*)lp = *(const v4f*)gp;
#endif
            }
        }
        // prefetch next K-tile of the weight stream (-> global_prefetch_b8)
        if (kt + 1 < kTiles) {
            int pn = nBlock + (tid & (GEMM_NT - 1));
            if (pn < N)
                __builtin_prefetch(&Wt[(size_t)pn * K + kBase + GEMM_KT], 0, 1);
        }
#if HAVE_ASYNC_LDS
        __builtin_amdgcn_s_wait_asynccnt(0);
#endif
        __syncthreads();

        // ---- 2 k-halves x 2 n-tiles per wave = 4 WMMAs ----
        int mrow = lane & 15;
        #pragma unroll
        for (int h = 0; h < 2; ++h) {
            // A fragment: elems 0..15 -> K kA..kA+15
            union { v4u u[2]; v16bf v; } au;
            int kA = h * 32 + ((lane < 16) ? 0 : 16);
            au.u[0] = *(const v4u*)&As[mrow][kA];
            au.u[1] = *(const v4u*)&As[mrow][kA + 8];
            int kB = h * 32 + ((lane < 16) ? 0 : 16);
            #pragma unroll
            for (int t = 0; t < 2; ++t) {
                int nrow = (wave * 2 + t) * 16 + (lane & 15);
                union { __bf16 e[16]; v16bf v; } bu;
                const v4f* wp = (const v4f*)&WsF[nrow][kB];
                #pragma unroll
                for (int q = 0; q < 4; ++q) {
                    v4f wv = wp[q];
                    bu.e[q * 4 + 0] = cvtbf(wv[0]);
                    bu.e[q * 4 + 1] = cvtbf(wv[1]);
                    bu.e[q * 4 + 2] = cvtbf(wv[2]);
                    bu.e[q * 4 + 3] = cvtbf(wv[3]);
                }
                acc[t] = __builtin_amdgcn_wmma_f32_16x16x32_bf16(
                             false, au.v, false, bu.v, (short)0, acc[t], false, false);
            }
        }
        __syncthreads();
    }

    // ---- store (16x16 f32 D layout: lane n = lane&15, VGPR r -> M=r(+8)) ----
    #pragma unroll
    for (int t = 0; t < 2; ++t) {
        int ncol = nBlock + (wave * 2 + t) * 16 + (lane & 15);
        if (ncol >= N) continue;
        float bv = bias ? bias[ncol] : 0.f;
        #pragma unroll
        for (int r = 0; r < 8; ++r) {
            int mr = m0 + r + ((lane < 16) ? 0 : 8);
            Cout[(size_t)mr * N + ncol] = acc[t][r] + bv;
        }
    }
}

// ---------------------------------------------------------------------------
// m3 scan + gate: thread per (b,d), 8 states in registers, L=3.
// ---------------------------------------------------------------------------
__global__ void m3_scan_gate_kernel(const float* __restrict__ delta_raw,
                                    const float* __restrict__ xdbl,
                                    const float* __restrict__ xh,
                                    const float* __restrict__ xz,
                                    const float* __restrict__ alog,
                                    const float* __restrict__ Dp,
                                    float* __restrict__ yg,
                                    int total) {
    int i = blockIdx.x * blockDim.x + threadIdx.x;
    if (i >= total) return;
    int d = i & 8191, b = i >> 13;
    float A[8], h[8];
    #pragma unroll
    for (int n = 0; n < 8; ++n) { A[n] = -__expf(alog[d * 8 + n]); h[n] = 0.f; }
    float Dd = Dp[d];
    #pragma unroll
    for (int l = 0; l < 3; ++l) {
        size_t row = (size_t)b * 3 + l;
        float delta = softplusf(delta_raw[row * 8192 + d]);
        float xv    = xh[row * 8192 + d];
        float y = 0.f;
        #pragma unroll
        for (int n = 0; n < 8; ++n) {
            float Bv = xdbl[row * 272 + 256 + n];
            float Cv = xdbl[row * 272 + 264 + n];
            h[n] = __expf(delta * A[n]) * h[n] + delta * Bv * xv;
            y += h[n] * Cv;
        }
        y += xv * Dd;
        float z = xz[row * 16384 + 8192 + d];
        yg[row * 8192 + d] = y * siluf(z);
    }
}

// ---------------------------------------------------------------------------
// softmax over batch axis of (B, 4096, 3)
// ---------------------------------------------------------------------------
__global__ void softmax_batch_kernel(const float* __restrict__ in,
                                     float* __restrict__ out,
                                     int inner, int Bn) {
    int i = blockIdx.x * blockDim.x + threadIdx.x;
    if (i >= inner) return;
    float mx = -3.4e38f;
    for (int b = 0; b < Bn; ++b) mx = fmaxf(mx, in[(size_t)b * inner + i]);
    float s = 0.f;
    for (int b = 0; b < Bn; ++b) s += __expf(in[(size_t)b * inner + i] - mx);
    float inv = 1.f / s;
    for (int b = 0; b < Bn; ++b)
        out[(size_t)b * inner + i] = __expf(in[(size_t)b * inner + i] - mx) * inv;
}

// prod[b, i, c] = att[b, i, c] * out3[b*3+c, i]
__global__ void att_mul_kernel(const float* __restrict__ att,
                               const float* __restrict__ out3,
                               float* __restrict__ prod, int total) {
    int idx = blockIdx.x * blockDim.x + threadIdx.x;
    if (idx >= total) return;
    int c = idx % 3;
    int i = (idx / 3) & 4095;
    int b = idx / 12288;
    prod[idx] = att[idx] * out3[((size_t)(b * 3 + c)) * 4096 + i];
}

// ---------------------------------------------------------------------------
// Host launcher
// ---------------------------------------------------------------------------
extern "C" void kernel_launch(void* const* d_in, const int* in_sizes, int n_in,
                              void* d_out, int out_size, void* d_ws, size_t ws_size,
                              hipStream_t stream) {
    (void)in_sizes; (void)n_in; (void)out_size; (void)ws_size;
    const int Bn = 32, L1 = 4096;
    const float* x        = (const float*)d_in[0];
    const float* conv1_w  = (const float*)d_in[1];
    const float* conv1_b  = (const float*)d_in[2];
    const float* conv2_w  = (const float*)d_in[3];
    const float* conv2_b  = (const float*)d_in[4];
    const float* ln_g     = (const float*)d_in[5];
    const float* ln_b     = (const float*)d_in[6];
    const float* smooth_w = (const float*)d_in[34];
    const float* smooth_b = (const float*)d_in[35];

    float* w = (float*)d_ws;
    size_t o = 0;
    auto alloc = [&](size_t n) { float* p = w + o; o += n; return p; };
    float* t    = alloc((size_t)Bn * 16 * 4096);
    float* xres = alloc((size_t)Bn * 3 * 4096);
    float* xs   = alloc((size_t)Bn * L1 * 3);
    float* xz1  = alloc((size_t)Bn * L1 * 12);
    float* xh1  = alloc((size_t)Bn * L1 * 6);
    float* xdb1 = alloc((size_t)Bn * L1 * 33);
    float* y1   = alloc((size_t)Bn * L1 * 6);
    float* out1 = alloc((size_t)Bn * L1 * 3);
    float* xz2  = alloc((size_t)Bn * L1 * 12);
    float* xh2  = alloc((size_t)Bn * L1 * 6);
    float* xdb2 = alloc((size_t)Bn * L1 * 33);
    float* y2   = alloc((size_t)Bn * L1 * 6);
    float* out2 = alloc((size_t)Bn * L1 * 3);
    float* att  = alloc((size_t)Bn * L1 * 3);
    float* xz3  = alloc((size_t)96 * 16384);
    float* xh3  = alloc((size_t)96 * 8192);
    float* xdb3 = alloc((size_t)96 * 272);
    float* del3 = alloc((size_t)96 * 8192);
    float* yg3  = alloc((size_t)96 * 8192);
    float* out3 = alloc((size_t)96 * 4096);
    float* prod = alloc((size_t)Bn * L1 * 3);

    const int TPB = 256;
    auto blks = [](int n) { return (n + 255) / 256; };

    conv3x3_kernel<<<blks(Bn * 16 * 4096), TPB, 0, stream>>>(
        x, conv1_w, conv1_b, nullptr, t, Bn, 3, 16, 1);
    conv3x3_kernel<<<blks(Bn * 3 * 4096), TPB, 0, stream>>>(
        t, conv2_w, conv2_b, x, xres, Bn, 16, 3, 0);
    layernorm_c3_kernel<<<blks(Bn * L1), TPB, 0, stream>>>(xres, ln_g, ln_b, xs, Bn * L1);

    auto run_small = [&](int base, const float* in, float* xz, float* xh,
                         float* xdb, float* y, float* outp) {
        const float* ip = (const float*)d_in[base + 0];
        const float* cw = (const float*)d_in[base + 1];
        const float* cb = (const float*)d_in[base + 2];
        const float* xp = (const float*)d_in[base + 3];
        const float* dw = (const float*)d_in[base + 4];
        const float* db = (const float*)d_in[base + 5];
        const float* al = (const float*)d_in[base + 6];
        const float* Dp = (const float*)d_in[base + 7];
        const float* op = (const float*)d_in[base + 8];
        sm_inproj_kernel<<<blks(Bn * L1), TPB, 0, stream>>>(in, ip, xz, Bn * L1);
        dwconv_silu_kernel<<<blks(Bn * L1 * 6), TPB, 0, stream>>>(xz, cw, cb, xh, Bn, L1, 6);
        sm_xproj_kernel<<<blks(Bn * L1), TPB, 0, stream>>>(xh, xp, xdb, Bn * L1);
        sm_scan_kernel<<<Bn * 6, 32, 0, stream>>>(xdb, xh, dw, db, al, Dp, y, L1);
        sm_gate_out_kernel<<<blks(Bn * L1), TPB, 0, stream>>>(y, xz, op, outp, Bn * L1);
    };
    run_small(7,  xs, xz1, xh1, xdb1, y1, out1);   // m1
    run_small(16, xs, xz2, xh2, xdb2, y2, out2);   // m2

    softmax_batch_kernel<<<blks(L1 * 3), TPB, 0, stream>>>(out2, att, L1 * 3, Bn);

    const float* ip3 = (const float*)d_in[25];
    const float* cw3 = (const float*)d_in[26];
    const float* cb3 = (const float*)d_in[27];
    const float* xp3 = (const float*)d_in[28];
    const float* dw3 = (const float*)d_in[29];
    const float* db3 = (const float*)d_in[30];
    const float* al3 = (const float*)d_in[31];
    const float* D3  = (const float*)d_in[32];
    const float* op3 = (const float*)d_in[33];

    {   // in_proj: (96,4096) x (16384,4096)^T, A gathered from out1
        dim3 g((16384 + GEMM_NT - 1) / GEMM_NT, 96 / 16);
        gemm_bf16_wmma_kernel<<<g, 128, 0, stream>>>(
            out1, 1, 0, ip3, nullptr, xz3, 96, 16384, 4096);
    }
    dwconv_silu_kernel<<<blks(Bn * 3 * 8192), TPB, 0, stream>>>(xz3, cw3, cb3, xh3, Bn, 3, 8192);
    {   // x_proj: (96,8192) x (272,8192)^T
        dim3 g((272 + GEMM_NT - 1) / GEMM_NT, 96 / 16);
        gemm_bf16_wmma_kernel<<<g, 128, 0, stream>>>(
            xh3, 0, 8192, xp3, nullptr, xdb3, 96, 272, 8192);
    }
    {   // dt_proj: A = xdb3[:, :256] (row stride 272), +bias
        dim3 g((8192 + GEMM_NT - 1) / GEMM_NT, 96 / 16);
        gemm_bf16_wmma_kernel<<<g, 128, 0, stream>>>(
            xdb3, 0, 272, dw3, db3, del3, 96, 8192, 256);
    }
    m3_scan_gate_kernel<<<blks(Bn * 8192), TPB, 0, stream>>>(
        del3, xdb3, xh3, xz3, al3, D3, yg3, Bn * 8192);
    {   // out_proj: (96,8192) x (4096,8192)^T
        dim3 g((4096 + GEMM_NT - 1) / GEMM_NT, 96 / 16);
        gemm_bf16_wmma_kernel<<<g, 128, 0, stream>>>(
            yg3, 0, 8192, op3, nullptr, out3, 96, 4096, 8192);
    }

    att_mul_kernel<<<blks(Bn * L1 * 3), TPB, 0, stream>>>(att, out3, prod, Bn * L1 * 3);

    conv3x3_kernel<<<blks(Bn * 3 * 4096), TPB, 0, stream>>>(
        prod, smooth_w, smooth_b, nullptr, (float*)d_out, Bn, 3, 3, 0);
}
